// TensHash_50362786513351
// MI455X (gfx1250) — 8-bit matrix operand order repaired and checked by reading, compiled, not run on hardware
//
#include <hip/hip_runtime.h>

typedef __attribute__((ext_vector_type(8))) int v8i;

#define H 256
#define RNDS 64
#define BATCH 65536
#define MAT_STRIDE 272
#define ST_STRIDE 272
#define MAT_BYTES (H * MAT_STRIDE)
#define STBUF_BYTES (16 * ST_STRIDE)
#define SMEM_BYTES (MAT_BYTES + 8 * STBUF_BYTES)

__device__ __forceinline__ int kbaseA(int v, int half) {
    return ((v >> 2) << 5) + (((v >> 1) & 1) << 4) + (half << 3) + ((v & 1) << 2);
}
__device__ __forceinline__ int kbaseB(int v, int half) {
    return ((v >> 2) << 5) + (((v >> 1) & 1) << 4) + (half << 3) + ((v & 1) << 2);
}

__global__ __launch_bounds__(256) void convert_mats_kernel(
    const float* __restrict__ mf, int* __restrict__ m8, int n4) {
    int i = blockIdx.x * blockDim.x + threadIdx.x;
    if (i < n4) {
        const float4 f = ((const float4*)mf)[i];
        int p = ((int)f.x & 255) | (((int)f.y & 255) << 8) |
                (((int)f.z & 255) << 16) | (((int)f.w & 255) << 24);
        m8[i] = p;
    }
}

__global__ __launch_bounds__(256) void tenshash_kernel(
    const float* __restrict__ state0,
    const float* __restrict__ matsF,
    const float* __restrict__ noiseF,
    const signed char* __restrict__ mats8,
    float* __restrict__ out,
    int use_ws)
{
    extern __shared__ char smem[];
    char* matB = smem;
    const int tid  = threadIdx.x;
    const int lane = tid & 31;
    const int wave = tid >> 5;
    const int half = lane >> 4;
    const int m    = lane & 15;
    char* stB = smem + MAT_BYTES + wave * STBUF_BYTES;
    const int rowBase = blockIdx.x * 128 + wave * 16;
    const unsigned matLds = (unsigned)(size_t)matB;

    int a_regs[32];
    {
        const float* srow = state0 + (size_t)(rowBase + m) * H;
#pragma unroll
        for (int c = 0; c < 4; ++c)
#pragma unroll
            for (int v = 0; v < 8; ++v) {
                int k = c * 64 + kbaseA(v, half);
                float4 f = *(const float4*)(srow + k);
                a_regs[c * 8 + v] = ((int)f.x & 255) | (((int)f.y & 255) << 8) |
                                    (((int)f.z & 255) << 16) | (((int)f.w & 255) << 24);
            }
    }

    for (int r = 0; r < RNDS; ++r) {
        __syncthreads();

        if (use_ws) {
            const int rbase = r * (H * H);
#pragma unroll
            for (int i = 0; i < 16; ++i) {
                int e = tid + i * 256;
                int row = e >> 4, ch = e & 15;
                unsigned voff = (unsigned)(rbase + row * H + ch * 16);
                unsigned loff = matLds + (unsigned)(row * MAT_STRIDE + ch * 16);
                asm volatile("global_load_async_to_lds_b128 %0, %1, %2"
                             :: "v"(loff), "v"(voff), "s"(mats8) : "memory");
            }
            asm volatile("s_wait_asynccnt 0" ::: "memory");
        } else {
            const float* mrf = matsF + (size_t)r * (H * H);
            for (int j = 0; j < 64; ++j) {
                int g = tid + j * 256;
                int row = g >> 6, cg = g & 63;
                float4 f = *(const float4*)(mrf + row * H + cg * 4);
                int p = ((int)f.x & 255) | (((int)f.y & 255) << 8) |
                        (((int)f.z & 255) << 16) | (((int)f.w & 255) << 24);
                *(int*)(matB + row * MAT_STRIDE + cg * 4) = p;
            }
        }
        __syncthreads();

        const float* nrow = noiseF + r * H;
        for (int nt = 0; nt < 16; ++nt) {
            const int col = nt * 16 + m;
            const int nz = (int)nrow[col];
            v8i acc = {};
#pragma unroll
            for (int c = 0; c < 4; ++c) {
                v8i a, b;
#pragma unroll
                for (int v = 0; v < 8; ++v) {
                    a[v] = a_regs[c * 8 + v];
                    b[v] = *(const int*)(matB + col * MAT_STRIDE + c * 64 + kbaseB(v, half));
                }
                acc = __builtin_amdgcn_wmma_i32_16x16x64_iu8(
                         true, a,  true, b, acc,
                         false,  false);
            }
#pragma unroll
            for (int d = 0; d < 8; ++d) {
                int t = acc[d] + nz;
                stB[(d + half * 8) * ST_STRIDE + col] = (signed char)(t % 2);
            }
        }

#pragma unroll
        for (int c = 0; c < 4; ++c)
#pragma unroll
            for (int v = 0; v < 8; ++v)
                a_regs[c * 8 + v] =
                    *(const int*)(stB + m * ST_STRIDE + c * 64 + kbaseA(v, half));
    }

    float* orow = out + (size_t)(rowBase + m) * H;
#pragma unroll
    for (int c = 0; c < 4; ++c)
#pragma unroll
        for (int v = 0; v < 8; ++v) {
            int k = c * 64 + kbaseA(v, half);
            int p = a_regs[c * 8 + v];
            float4 f;
            f.x = (float)(signed char)(p & 255);
            f.y = (float)(signed char)((p >> 8) & 255);
            f.z = (float)(signed char)((p >> 16) & 255);
            f.w = (float)(signed char)((p >> 24) & 255);
            *(float4*)(orow + k) = f;
        }
}

extern "C" void kernel_launch(void* const* d_in, const int* in_sizes, int n_in,
                              void* d_out, int out_size, void* d_ws, size_t ws_size,
                              hipStream_t stream) {
    const float* state = (const float*)d_in[0];
    const float* mats  = (const float*)d_in[1];
    const float* noise = (const float*)d_in[2];
    float* out = (float*)d_out;

    const size_t mat8_bytes = (size_t)RNDS * H * H;
    int use_ws = (ws_size >= mat8_bytes) ? 1 : 0;

    if (use_ws) {
        const int n4 = RNDS * H * H / 4;
        convert_mats_kernel<<<(n4 + 255) / 256, 256, 0, stream>>>(mats, (int*)d_ws, n4);
    }
    tenshash_kernel<<<BATCH / 128, 256, SMEM_BYTES, stream>>>(
        state, mats, noise, (const signed char*)d_ws, out, use_ws);
}
